// FeatureGATLayer_61916248539242
// MI455X (gfx1250) — matrix operand order repaired and checked by reading, not compiled, not run on hardware
//
#include <hip/hip_runtime.h>

typedef __attribute__((ext_vector_type(16))) _Float16     v16h;
typedef __attribute__((ext_vector_type(8)))  float        v8f;
typedef __attribute__((ext_vector_type(4)))  unsigned int v4u;
typedef __attribute__((ext_vector_type(4)))  int          v4i_t;

#define GAT_ALPHA 0.2f

union V16H { v16h h; v4u u[2]; };
union H8   { _Float16 f[8]; v4u u; };

static __device__ __forceinline__ float lrelu(float z) {
  // alpha < 1 => leaky_relu(z) == max(z, alpha*z) for all z
  return fmaxf(z, GAT_ALPHA * z);
}

// Async global->LDS direct copies (gfx1250 GLOBAL_LOAD_ASYNC_TO_LDS_B128),
// with a plain load + ds_store fallback if the builtin is unavailable.
// Builtin signature (from hipcc diagnostic): (v4i* src, v4i* lds_dst, Ii off, Ii cpol)
#if defined(__AMDGCN__) && __has_builtin(__builtin_amdgcn_global_load_async_to_lds_b128)
#define HAVE_ASYNC_LDS 1
typedef __attribute__((address_space(1))) v4i_t* gv4i_p;
typedef __attribute__((address_space(3))) v4i_t* lv4i_p;
// OFF must be a literal (IOFFSET immediate applies to both addresses per ISA)
#define ASYNC_CP_B128(gptr, lptr, OFF)                                        \
  __builtin_amdgcn_global_load_async_to_lds_b128(                             \
      (gv4i_p)(uintptr_t)(gptr), (lv4i_p)(unsigned int)(uintptr_t)(lptr),     \
      OFF, 0)
static __device__ __forceinline__ void async_wait0() {
#if __has_builtin(__builtin_amdgcn_s_wait_asynccnt)
  __builtin_amdgcn_s_wait_asynccnt(0);
#else
  asm volatile("s_wait_asynccnt 0x0" ::: "memory");
#endif
}
#endif

// ---------------------------------------------------------------------------
// Kernel 0a: x (f32, row-major [b][n][k]) -> x16 (f16, same layout)
// ---------------------------------------------------------------------------
__global__ void k_cvt_x(const float* __restrict__ x, _Float16* __restrict__ x16, int n) {
  for (int i = blockIdx.x * blockDim.x + threadIdx.x; i < n; i += gridDim.x * blockDim.x)
    x16[i] = (_Float16)x[i];
}

// ---------------------------------------------------------------------------
// Kernel 0b: W (f32, [k][d] 128x128) -> W16T (f16, [d][k]) (B-tile friendly)
// ---------------------------------------------------------------------------
__global__ void k_cvt_w(const float* __restrict__ W, _Float16* __restrict__ W16T) {
  int idx = blockIdx.x * 256 + threadIdx.x;    // 0..16383
  int d = idx >> 7, k = idx & 127;
  W16T[d * 128 + k] = (_Float16)W[k * 128 + d];
}

// ---------------------------------------------------------------------------
// Kernel 1: h = x @ W via v_wmma_f32_16x16x32_f16.
// One wave = 16 rows x 128 cols of h. Writes h16T[b][d][j] (f16, transposed)
// and s1/s2 (f32) computed from the C registers.
// ---------------------------------------------------------------------------
__global__ void __launch_bounds__(256) k_gemm_h(
    const _Float16* __restrict__ x16, const _Float16* __restrict__ W16T,
    const float* __restrict__ a, _Float16* __restrict__ h16T,
    float* __restrict__ s1, float* __restrict__ s2)
{
  const int lane = threadIdx.x & 31;
  const int wave = threadIdx.x >> 5;
  const int gw   = blockIdx.x * 8 + wave;   // 0..2047 row-blocks
  const int b    = gw >> 7;                 // batch
  const int i0   = (gw & 127) << 4;         // first row of 16-row block
  const int half = lane >> 4;
  const int l15  = lane & 15;

  v8f c[8] = {};                            // 16x128 f32 accumulators

  const _Float16* xrow = x16 + (size_t)(b * 2048 + i0 + l15) * 128;

  #pragma unroll
  for (int k0 = 0; k0 < 128; k0 += 32) {
    V16H A;                                 // 16x32 A tile of x
    A.u[0] = *(const v4u*)(xrow + k0 + half * 16);
    A.u[1] = *(const v4u*)(xrow + k0 + half * 16 + 8);
    #pragma unroll
    for (int t = 0; t < 8; ++t) {
      const _Float16* wp = W16T + (size_t)(t * 16 + l15) * 128 + k0 + half * 16;
      V16H B;                               // 32x16 B tile of W
      B.u[0] = ((const v4u*)wp)[0];
      B.u[1] = ((const v4u*)wp)[1];
      c[t] = __builtin_amdgcn_wmma_f32_16x16x32_f16(false, A.h, false, B.h,
                                                    (short)0, c[t], false, false);
    }
  }

  // Store h16T[b][d][j]: lane owns column d = t*16+l15, rows j = i0+half*8+r
  #pragma unroll
  for (int t = 0; t < 8; ++t) {
    H8 o;
    #pragma unroll
    for (int r = 0; r < 8; ++r) o.f[r] = (_Float16)c[t][r];
    *(v4u*)(h16T + (size_t)(b * 128 + t * 16 + l15) * 2048 + i0 + half * 8) = o.u;
  }

  // s1/s2: dot each h-row with a1/a2 straight from C registers
  float a1v[8], a2v[8];
  #pragma unroll
  for (int t = 0; t < 8; ++t) {
    a1v[t] = a[t * 16 + l15];
    a2v[t] = a[128 + t * 16 + l15];
  }
  #pragma unroll
  for (int r = 0; r < 8; ++r) {
    float p1 = 0.f, p2 = 0.f;
    #pragma unroll
    for (int t = 0; t < 8; ++t) { p1 += c[t][r] * a1v[t]; p2 += c[t][r] * a2v[t]; }
    #pragma unroll
    for (int m = 1; m <= 8; m <<= 1) {       // reduce across the 16 columns
      p1 += __shfl_xor(p1, m, 32);
      p2 += __shfl_xor(p2, m, 32);
    }
    if (l15 == 0) {
      s1[b * 2048 + i0 + half * 8 + r] = p1;
      s2[b * 2048 + i0 + half * 8 + r] = p2;
    }
  }
}

// ---------------------------------------------------------------------------
// Kernel 2: m2[b] = max_j s2[b][j]  (softmax row-max via lrelu monotonicity)
// ---------------------------------------------------------------------------
__global__ void k_rowmax(const float* __restrict__ s2, float* __restrict__ m2) {
  __shared__ float red[256];
  const int b = blockIdx.x;
  float m = -3.0e38f;
  for (int j = threadIdx.x; j < 2048; j += 256) m = fmaxf(m, s2[b * 2048 + j]);
  red[threadIdx.x] = m;
  __syncthreads();
  for (int s = 128; s > 0; s >>= 1) {
    if (threadIdx.x < s) red[threadIdx.x] = fmaxf(red[threadIdx.x], red[threadIdx.x + s]);
    __syncthreads();
  }
  if (threadIdx.x == 0) m2[b] = red[0];
}

// ---------------------------------------------------------------------------
// Kernel 3: fused attention: out = relu( softmax(lrelu(s1_i+s2_j)) @ h )
// One wave = 16 output rows; single sweep over j (row-max known a priori).
// The 32j x 128d h16T block is staged in LDS (double buffered) and shared by
// all 8 waves of the block -> 8x less L2 traffic. A = exp-weights built
// in-register, B = LDS tiles via ds_load_b128, f32 accumulate.
// ---------------------------------------------------------------------------
#define JSTRIDE 40  // f16 per LDS row: 32 data + 8 pad (20 dwords, bank-friendly)

__global__ void __launch_bounds__(256) k_attn(
    const _Float16* __restrict__ h16T, const float* __restrict__ s1,
    const float* __restrict__ s2, const float* __restrict__ m2,
    float* __restrict__ out)
{
  __shared__ float   s2sh[2048];              // 8 KB
  __shared__ _Float16 hsh[2][128 * JSTRIDE];  // 2 x 10 KB

  const int tid  = threadIdx.x;
  const int lane = tid & 31;
  const int wave = tid >> 5;
  const int gw   = blockIdx.x * 8 + wave;
  const int b    = gw >> 7;
  const int i0   = (gw & 127) << 4;
  const int half = lane >> 4;
  const int l15  = lane & 15;

  for (int j = tid; j < 2048; j += 256) s2sh[j] = s2[b * 2048 + j];

  // staging role: thread tid moves 32B: row d = tid>>1, j-half = (tid&1)*16
  const int sd  = tid >> 1;
  const int sj  = (tid & 1) * 16;
  const _Float16* gstage = h16T + (size_t)(b * 128 + sd) * 2048 + sj;
  _Float16* lst0 = &hsh[0][sd * JSTRIDE + sj];
  _Float16* lst1 = &hsh[1][sd * JSTRIDE + sj];

#if HAVE_ASYNC_LDS
  ASYNC_CP_B128(gstage, lst0, 0);              // stage 0 (j0 = 0)
  ASYNC_CP_B128(gstage, lst0, 16);
  async_wait0();
#else
  v4u r0 = ((const v4u*)gstage)[0];
  v4u r1 = ((const v4u*)gstage)[1];
  ((v4u*)lst0)[0] = r0;
  ((v4u*)lst0)[1] = r1;
#endif
  __syncthreads();

  const float s1r = s1[b * 2048 + i0 + l15];   // this lane's A-row
  const float mi  = lrelu(s1r + m2[b]);        // exact softmax max for row

  v8f  c[8] = {};
  float denom = 0.f;

  for (int it = 0; it < 64; ++it) {
    const int j0  = it * 32;
    _Float16* cbuf = (it & 1) ? &hsh[1][0] : &hsh[0][0];
    _Float16* nst  = (it & 1) ? lst0 : lst1;

    // kick off next stage (overlaps with the WMMAs below)
#if HAVE_ASYNC_LDS
    if (it + 1 < 64) {
      ASYNC_CP_B128(gstage + (j0 + 32), nst, 0);
      ASYNC_CP_B128(gstage + (j0 + 32), nst, 16);
    }
#else
    v4u p0, p1;
    if (it + 1 < 64) {
      p0 = ((const v4u*)(gstage + (j0 + 32)))[0];
      p1 = ((const v4u*)(gstage + (j0 + 32)))[1];
    }
#endif

    // A tile: 16x32 attention weights, exact f32 softmax math
    V16H A;
    #pragma unroll
    for (int q = 0; q < 2; ++q) {
      const int jb = j0 + half * 16 + q * 8;
      #pragma unroll
      for (int cc = 0; cc < 8; ++cc) {
        const float z = s1r + s2sh[jb + cc];
        const float w = __expf(lrelu(z) - mi);   // <= 1
        denom += w;
        A.h[q * 8 + cc] = (_Float16)w;
      }
    }

    // B tiles from LDS + 8 WMMAs
    #pragma unroll
    for (int t = 0; t < 8; ++t) {
      const _Float16* hp = cbuf + (t * 16 + l15) * JSTRIDE + half * 16;
      V16H B;
      B.u[0] = ((const v4u*)hp)[0];
      B.u[1] = ((const v4u*)hp)[1];
      c[t] = __builtin_amdgcn_wmma_f32_16x16x32_f16(false, A.h, false, B.h,
                                                    (short)0, c[t], false, false);
    }

#if HAVE_ASYNC_LDS
    if (it + 1 < 64) async_wait0();
#else
    if (it + 1 < 64) {
      ((v4u*)nst)[0] = p0;
      ((v4u*)nst)[1] = p1;
    }
#endif
    __syncthreads();   // publish next buffer; cur buffer free to overwrite
  }

  // full row denominators: combine the two lane-halves, then broadcast
  denom += __shfl_xor(denom, 16, 32);
  float dinv[8];
  #pragma unroll
  for (int r = 0; r < 8; ++r)
    dinv[r] = 1.0f / __shfl(denom, r + half * 8, 32);

  float* ob = out + (size_t)b * 2048 * 128;
  #pragma unroll
  for (int t = 0; t < 8; ++t) {
    #pragma unroll
    for (int r = 0; r < 8; ++r) {
      const float v = c[t][r] * dinv[r];
      ob[(size_t)(i0 + half * 8 + r) * 128 + t * 16 + l15] = fmaxf(v, 0.f);
    }
  }
}

// ---------------------------------------------------------------------------
// Workspace layout (bytes):
//   x16   @ 0         : 16*2048*128*2 = 8388608
//   h16T  @ 8388608   : 16*128*2048*2 = 8388608
//   W16T  @ 16777216  : 128*128*2     = 32768
//   s1    @ 16809984  : 16*2048*4     = 131072
//   s2    @ 16941056  : 16*2048*4     = 131072
//   m2    @ 17072128  : 16*4
// ---------------------------------------------------------------------------
extern "C" void kernel_launch(void* const* d_in, const int* in_sizes, int n_in,
                              void* d_out, int out_size, void* d_ws, size_t ws_size,
                              hipStream_t stream) {
  (void)in_sizes; (void)n_in; (void)out_size; (void)ws_size;
  const float* x = (const float*)d_in[0];
  const float* W = (const float*)d_in[1];
  const float* a = (const float*)d_in[2];
  float* out = (float*)d_out;

  char* ws = (char*)d_ws;
  _Float16* x16  = (_Float16*)(ws);
  _Float16* h16T = (_Float16*)(ws + 8388608);
  _Float16* W16T = (_Float16*)(ws + 16777216);
  float* s1 = (float*)(ws + 16809984);
  float* s2 = (float*)(ws + 16941056);
  float* m2 = (float*)(ws + 17072128);

  k_cvt_x<<<4096, 256, 0, stream>>>(x, x16, 16 * 2048 * 128);
  k_cvt_w<<<64, 256, 0, stream>>>(W, W16T);
  k_gemm_h<<<256, 256, 0, stream>>>(x16, W16T, a, h16T, s1, s2);
  k_rowmax<<<16, 256, 0, stream>>>(s2, m2);
  k_attn<<<256, 256, 0, stream>>>(h16T, s1, s2, m2, out);
}
